// GroupingBlock_223338299412
// MI455X (gfx1250) — matrix operand order repaired and checked by reading, not compiled, not run on hardware
//
#include <hip/hip_runtime.h>

// ---------------------------------------------------------------------------
// CDNA5 / gfx1250 WMMA types and helpers (wave32)
// ---------------------------------------------------------------------------
typedef __attribute__((ext_vector_type(16))) _Float16 v16h;
typedef __attribute__((ext_vector_type(8)))  float    v8f;

#define WMMA16(a, b, c) \
  __builtin_amdgcn_wmma_f32_16x16x32_f16(false, (a), false, (b), (short)0, (c), false, false)

union AFrag { v16h v; uint4 q[2]; };
union BFrag { v16h v; unsigned int u[8]; };

__device__ __forceinline__ float gelu_f(float x) {
  return 0.5f * x * (1.0f + erff(x * 0.70710678118654752f));
}

// block-wide sum, blockDim.x == 256 (8 waves of 32)
__device__ __forceinline__ float blk_sum(float v, float* sbuf) {
  __syncthreads();
#pragma unroll
  for (int o = 16; o > 0; o >>= 1) v += __shfl_xor(v, o, 32);
  const int lane = threadIdx.x & 31, wid = threadIdx.x >> 5;
  if (lane == 0) sbuf[wid] = v;
  __syncthreads();
  if (threadIdx.x == 0) {
    float s = 0.f;
    for (int i = 0; i < 8; i++) s += sbuf[i];
    sbuf[0] = s;
  }
  __syncthreads();
  return sbuf[0];
}

// ---------------------------------------------------------------------------
// Elementwise utility kernels
// ---------------------------------------------------------------------------
__global__ void k_cvt16(const float* __restrict__ in, _Float16* __restrict__ out, long long n) {
  long long i = (long long)blockIdx.x * blockDim.x + threadIdx.x;
  long long st = (long long)gridDim.x * blockDim.x;
  for (; i < n; i += st) out[i] = (_Float16)in[i];
}

// out is [cols_in, rows_in]; out[c*rows_in + r] = in[r*cols_in + c]
__global__ void k_transpose16(const float* __restrict__ in, _Float16* __restrict__ out,
                              int rows_in, int cols_in) {
  long long n = (long long)rows_in * cols_in;
  long long i = (long long)blockIdx.x * blockDim.x + threadIdx.x;
  long long st = (long long)gridDim.x * blockDim.x;
  for (; i < n; i += st) {
    int c = (int)(i / rows_in);
    int r = (int)(i % rows_in);
    out[i] = (_Float16)in[(size_t)r * cols_in + c];
  }
}

__global__ void k_zero(float* __restrict__ p, long long n) {
  long long i = (long long)blockIdx.x * blockDim.x + threadIdx.x;
  long long st = (long long)gridDim.x * blockDim.x;
  for (; i < n; i += st) p[i] = 0.f;
}

// ---------------------------------------------------------------------------
// LayerNorm kernels (feature dim fixed at 768 = 3*256)
// ---------------------------------------------------------------------------
__global__ __launch_bounds__(256) void k_ln768(const float* __restrict__ in,
                                               const float* __restrict__ g,
                                               const float* __restrict__ bt,
                                               _Float16* __restrict__ o16,
                                               float* __restrict__ o32) {
  __shared__ float sbuf[8];
  const size_t row = blockIdx.x;
  const float* xr = in + row * 768;
  float loc[3];
#pragma unroll
  for (int j = 0; j < 3; j++) loc[j] = xr[threadIdx.x + j * 256];
  float mean = blk_sum(loc[0] + loc[1] + loc[2], sbuf) * (1.f / 768.f);
  float s2 = 0.f;
#pragma unroll
  for (int j = 0; j < 3; j++) { float d = loc[j] - mean; s2 += d * d; }
  float var = blk_sum(s2, sbuf) * (1.f / 768.f);
  float rr = rsqrtf(var + 1e-5f);
#pragma unroll
  for (int j = 0; j < 3; j++) {
    int c = threadIdx.x + j * 256;
    float v = (loc[j] - mean) * rr * g[c] + bt[c];
    if (o16) o16[row * 768 + c] = (_Float16)v;
    if (o32) o32[row * 768 + c] = v;
  }
}

// LN rows of group_tokens [B,128,768], write transposed f16 A-matrix [B,768,128]
__global__ __launch_bounds__(256) void k_ln_gt_T(const float* __restrict__ gt,
                                                 const float* __restrict__ g,
                                                 const float* __restrict__ bt,
                                                 _Float16* __restrict__ outT) {
  __shared__ float sbuf[8];
  const int gi = blockIdx.x;  // 0..127
  const int b = blockIdx.y;
  const float* xr = gt + ((size_t)b * 128 + gi) * 768;
  float loc[3];
#pragma unroll
  for (int j = 0; j < 3; j++) loc[j] = xr[threadIdx.x + j * 256];
  float mean = blk_sum(loc[0] + loc[1] + loc[2], sbuf) * (1.f / 768.f);
  float s2 = 0.f;
#pragma unroll
  for (int j = 0; j < 3; j++) { float d = loc[j] - mean; s2 += d * d; }
  float var = blk_sum(s2, sbuf) * (1.f / 768.f);
  float rr = rsqrtf(var + 1e-5f);
#pragma unroll
  for (int j = 0; j < 3; j++) {
    int c = threadIdx.x + j * 256;
    float v = (loc[j] - mean) * rr * g[c] + bt[c];
    outT[((size_t)b * 768 + c) * 128 + gi] = (_Float16)v;
  }
}

// LN over C of t2^T:  t2 is [B,768,64];  pgt[b,gi,c] = LN_c(t2[b,c,gi])
__global__ __launch_bounds__(256) void k_ln_t2T(const float* __restrict__ t2,
                                                const float* __restrict__ g,
                                                const float* __restrict__ bt,
                                                _Float16* __restrict__ o16,
                                                float* __restrict__ o32) {
  __shared__ float sbuf[8];
  const int gi = blockIdx.x;  // 0..63
  const int b = blockIdx.y;
  float loc[3];
#pragma unroll
  for (int j = 0; j < 3; j++) {
    int c = threadIdx.x + j * 256;
    loc[j] = t2[((size_t)b * 768 + c) * 64 + gi];
  }
  float mean = blk_sum(loc[0] + loc[1] + loc[2], sbuf) * (1.f / 768.f);
  float s2 = 0.f;
#pragma unroll
  for (int j = 0; j < 3; j++) { float d = loc[j] - mean; s2 += d * d; }
  float var = blk_sum(s2, sbuf) * (1.f / 768.f);
  float rr = rsqrtf(var + 1e-5f);
#pragma unroll
  for (int j = 0; j < 3; j++) {
    int c = threadIdx.x + j * 256;
    float v = (loc[j] - mean) * rr * g[c] + bt[c];
    size_t idx = ((size_t)b * 64 + gi) * 768 + c;
    o16[idx] = (_Float16)v;
    o32[idx] = v;
  }
}

// out[row] = sum_c a16[row,c] * v[c]   (C = 768)
__global__ __launch_bounds__(256) void k_rowdot768(const _Float16* __restrict__ a,
                                                   const float* __restrict__ v,
                                                   float* __restrict__ out) {
  __shared__ float sbuf[8];
  const size_t row = blockIdx.x;
  float loc = 0.f;
#pragma unroll
  for (int j = 0; j < 3; j++) {
    int c = threadIdx.x + j * 256;
    loc += (float)a[row * 768 + c] * v[c];
  }
  float s = blk_sum(loc, sbuf);
  if (threadIdx.x == 0) out[row] = s;
}

// ---------------------------------------------------------------------------
// Generic batched WMMA GEMM.
//   A  : f16 [batch, M, K] row-major
//   Bw : f16 [K, N] row-major (shared across batch)
// B tile is restaged K-paired: BsP[k/2][n] = {B[2k][n], B[2k+1][n]} so B
// fragments are 8x ds_load_b32; A fragments are 2x ds_load_b128.
// ---------------------------------------------------------------------------
__global__ __launch_bounds__(256) void k_gemm(const _Float16* __restrict__ A,
                                              const _Float16* __restrict__ Bw,
                                              const float* __restrict__ bias,
                                              const float* __restrict__ res32,
                                              float* __restrict__ out32,
                                              _Float16* __restrict__ out16,
                                              const float* __restrict__ rowCnt,
                                              int M, int N, int K, float alpha, int doGelu) {
  __shared__ __align__(16) _Float16 As[64][40];
  __shared__ __align__(16) unsigned int BsP[16][68];
  const int bz = blockIdx.z;
  const int n0 = blockIdx.x * 64, m0 = blockIdx.y * 64;
  const _Float16* Ab = A + (size_t)bz * M * K;
  const int tid = threadIdx.x, lane = tid & 31, wave = tid >> 5;
  const int tm = wave >> 1;          // 0..3 : 16-row group
  const int tn0 = (wave & 1) * 2;    // {0,2}: pair of 16-col groups
  const int ar = tid >> 2, ac = (tid & 3) * 8;   // A staging: 64x32
  const int kp = tid >> 4, nq = (tid & 15) * 4;  // B staging: 16 k-pairs x 64 n
  v8f acc[2] = {};
  for (int kk = 0; kk < K; kk += 32) {
    {
      *(uint4*)&As[ar][ac] = *(const uint4*)(Ab + (size_t)(m0 + ar) * K + kk + ac);
      union { uint2 v; unsigned short s[4]; } r0, r1;
      r0.v = *(const uint2*)(Bw + (size_t)(kk + 2 * kp) * N + n0 + nq);
      r1.v = *(const uint2*)(Bw + (size_t)(kk + 2 * kp + 1) * N + n0 + nq);
      uint4 p4;
      p4.x = (unsigned)r0.s[0] | ((unsigned)r1.s[0] << 16);
      p4.y = (unsigned)r0.s[1] | ((unsigned)r1.s[1] << 16);
      p4.z = (unsigned)r0.s[2] | ((unsigned)r1.s[2] << 16);
      p4.w = (unsigned)r0.s[3] | ((unsigned)r1.s[3] << 16);
      *(uint4*)&BsP[kp][nq] = p4;
      if (kk + 32 < K) {
        __builtin_prefetch(Ab + (size_t)(m0 + ar) * K + kk + 32 + ac, 0, 0);
        __builtin_prefetch(Bw + (size_t)(kk + 32 + 2 * kp) * N + n0 + nq, 0, 0);
      }
    }
    __syncthreads();
    AFrag af;
    const int am = tm * 16 + (lane & 15);
    const int kof = (lane >> 4) * 16;
    af.q[0] = *(const uint4*)&As[am][kof];
    af.q[1] = *(const uint4*)&As[am][kof + 8];
    const int bn = lane & 15;
    const int kp0 = (lane >> 4) * 8;
    BFrag bf0, bf1;
#pragma unroll
    for (int j = 0; j < 8; j++) {
      bf0.u[j] = BsP[kp0 + j][tn0 * 16 + bn];
      bf1.u[j] = BsP[kp0 + j][(tn0 + 1) * 16 + bn];
    }
    acc[0] = WMMA16(af.v, bf0.v, acc[0]);
    acc[1] = WMMA16(af.v, bf1.v, acc[1]);
    __syncthreads();
  }
#pragma unroll
  for (int st = 0; st < 2; st++) {
    const int tn = tn0 + st;
#pragma unroll
    for (int r = 0; r < 8; r++) {
      const int ml = tm * 16 + r + ((lane >> 4) << 3);
      const int nl = tn * 16 + (lane & 15);
      const int gm = m0 + ml, gn = n0 + nl;
      const size_t row = (size_t)bz * M + gm;
      const size_t idx = row * N + gn;
      float v = acc[st][r];
      if (bias) {
        if (rowCnt) {
          float cn = rowCnt[row];
          v = (v + bias[gn] * cn) * (1.f / (cn + 1.f));
        } else {
          v += bias[gn];
        }
      }
      v *= alpha;
      if (doGelu) v = gelu_f(v);
      if (res32) v += res32[idx];
      if (out32) out32[idx] = v;
      if (out16) out16[idx] = (_Float16)v;
    }
  }
}

// ---------------------------------------------------------------------------
// Fused flash cross-attention, one block per (head, batch).
// K/V head slices are computed in-kernel with WMMA.  V is kept K(n)-paired so
// the P@V B-fragments are ds_load_b32; Q/P/A fragments are ds_load_b128.
// ---------------------------------------------------------------------------
__global__ __launch_bounds__(256) void k_flash(const _Float16* __restrict__ q16,
                                               const _Float16* __restrict__ xn16,
                                               const _Float16* __restrict__ wk,
                                               const _Float16* __restrict__ wv,
                                               const float* __restrict__ kb,
                                               const float* __restrict__ vb,
                                               _Float16* __restrict__ o16) {
  __shared__ __align__(16) _Float16 qs[64][72];
  __shared__ __align__(16) _Float16 a_sub[64][40];
  __shared__ __align__(16) unsigned int wkP[16][68];
  __shared__ __align__(16) unsigned int wvP[16][68];
  __shared__ __align__(16) _Float16 kT[64][72];   // K^T [d][n]; reused as P [q][n]
  __shared__ __align__(16) unsigned int vsP[32][68];  // V, n-paired: [n/2][d]
  __shared__ float sc[64][65];
  __shared__ float m_s[64], l_s[64], corr_s[64];

  const int h = blockIdx.x, b = blockIdx.y;
  const int d0 = h * 64;
  const int tid = threadIdx.x, lane = tid & 31, wave = tid >> 5;

  {  // load q tile [64 x 64]
    const int r = tid >> 2, c0 = (tid & 3) * 16;
    const _Float16* src = q16 + ((size_t)b * 64 + r) * 768 + d0 + c0;
    *(uint4*)&qs[r][c0] = *(const uint4*)src;
    *(uint4*)&qs[r][c0 + 8] = *(const uint4*)(src + 8);
  }
  if (tid < 64) { m_s[tid] = -1e30f; l_s[tid] = 0.f; }

  v8f of0 = {}, of1 = {};
  const int so = wave * 2;
  const int otm = so >> 2;     // q-row group for this wave's score/o tiles
  const int otn0 = so & 3;     // col group pair {0,2}
  const int pw = wave & 3;     // token-row group for the projection phase
  const bool isV = wave >= 4;
  __syncthreads();

  for (int nt = 0; nt < 4096 / 64; nt++) {
    const int n0 = nt * 64;
    v8f pk[4] = {};
    // ---- phase A: project K and V tiles (K = 768) ----
    for (int kk = 0; kk < 768; kk += 32) {
      {
        const int r = tid >> 2, c0 = (tid & 3) * 8;
        *(uint4*)&a_sub[r][c0] =
            *(const uint4*)(xn16 + ((size_t)b * 4096 + n0 + r) * 768 + kk + c0);
        const int kpp = tid >> 4, dq = (tid & 15) * 4;
        union { uint2 v; unsigned short s[4]; } r0, r1;
        uint4 p4;
        r0.v = *(const uint2*)(wk + (size_t)(kk + 2 * kpp) * 768 + d0 + dq);
        r1.v = *(const uint2*)(wk + (size_t)(kk + 2 * kpp + 1) * 768 + d0 + dq);
        p4.x = (unsigned)r0.s[0] | ((unsigned)r1.s[0] << 16);
        p4.y = (unsigned)r0.s[1] | ((unsigned)r1.s[1] << 16);
        p4.z = (unsigned)r0.s[2] | ((unsigned)r1.s[2] << 16);
        p4.w = (unsigned)r0.s[3] | ((unsigned)r1.s[3] << 16);
        *(uint4*)&wkP[kpp][dq] = p4;
        r0.v = *(const uint2*)(wv + (size_t)(kk + 2 * kpp) * 768 + d0 + dq);
        r1.v = *(const uint2*)(wv + (size_t)(kk + 2 * kpp + 1) * 768 + d0 + dq);
        p4.x = (unsigned)r0.s[0] | ((unsigned)r1.s[0] << 16);
        p4.y = (unsigned)r0.s[1] | ((unsigned)r1.s[1] << 16);
        p4.z = (unsigned)r0.s[2] | ((unsigned)r1.s[2] << 16);
        p4.w = (unsigned)r0.s[3] | ((unsigned)r1.s[3] << 16);
        *(uint4*)&wvP[kpp][dq] = p4;
      }
      __syncthreads();
      AFrag af;
      const int am = pw * 16 + (lane & 15);
      const int kof = (lane >> 4) * 16;
      af.q[0] = *(const uint4*)&a_sub[am][kof];
      af.q[1] = *(const uint4*)&a_sub[am][kof + 8];
      const int bn = lane & 15;
      const int kp0 = (lane >> 4) * 8;
#pragma unroll
      for (int t = 0; t < 4; t++) {
        BFrag bf;
#pragma unroll
        for (int j = 0; j < 8; j++)
          bf.u[j] = isV ? wvP[kp0 + j][t * 16 + bn] : wkP[kp0 + j][t * 16 + bn];
        pk[t] = WMMA16(af.v, bf.v, pk[t]);
      }
      __syncthreads();
    }
    // write K^T (b16 scatter) / V (n-paired b32) tiles to LDS with bias
#pragma unroll
    for (int t = 0; t < 4; t++) {
      const int dl = t * 16 + (lane & 15);
      if (!isV) {
        const float bb = kb[d0 + dl];
#pragma unroll
        for (int r = 0; r < 8; r++) {
          const int nl = pw * 16 + r + ((lane >> 4) << 3);
          kT[dl][nl] = (_Float16)(pk[t][r] + bb);
        }
      } else {
        const float bb = vb[d0 + dl];
#pragma unroll
        for (int r = 0; r < 8; r += 2) {
          const int nl = pw * 16 + r + ((lane >> 4) << 3);
          union { _Float16 hh[2]; unsigned u; } pr;
          pr.hh[0] = (_Float16)(pk[t][r] + bb);
          pr.hh[1] = (_Float16)(pk[t][r + 1] + bb);
          vsP[nl >> 1][dl] = pr.u;
        }
      }
    }
    __syncthreads();
    // ---- phase B: scores = q @ K^T  (K = 64) ----
    {
      v8f sacc[2] = {};
      for (int k2 = 0; k2 < 64; k2 += 32) {
        AFrag af;
        const int am = otm * 16 + (lane & 15);
        const int kof = (lane >> 4) * 16;
        af.q[0] = *(const uint4*)&qs[am][k2 + kof];
        af.q[1] = *(const uint4*)&qs[am][k2 + kof + 8];
        const int bn = lane & 15;
        const int kb0 = (lane >> 4) << 4;
#pragma unroll
        for (int st = 0; st < 2; st++) {
          v16h bf;
#pragma unroll
          for (int i = 0; i < 16; i++) bf[i] = kT[k2 + kb0 + i][(otn0 + st) * 16 + bn];
          sacc[st] = WMMA16(af.v, bf, sacc[st]);
        }
      }
#pragma unroll
      for (int st = 0; st < 2; st++)
#pragma unroll
        for (int r = 0; r < 8; r++)
          sc[otm * 16 + r + ((lane >> 4) << 3)][(otn0 + st) * 16 + (lane & 15)] = sacc[st][r];
    }
    __syncthreads();
    // ---- phase C: online softmax (64 rows) ----
    if (tid < 64) {
      float mx = -1e30f;
      for (int j = 0; j < 64; j++) mx = fmaxf(mx, sc[tid][j]);
      const float mold = m_s[tid];
      const float mnew = fmaxf(mold, mx);
      const float corr = __expf(mold - mnew);
      float ls = 0.f;
      for (int j = 0; j < 64; j++) {
        float p = __expf(sc[tid][j] - mnew);
        ls += p;
        kT[tid][j] = (_Float16)p;  // reuse kT as P[q][n]
      }
      l_s[tid] = l_s[tid] * corr + ls;
      m_s[tid] = mnew;
      corr_s[tid] = corr;
    }
    __syncthreads();
    // ---- phase D: o = o*corr + P @ V ----
    {
#pragma unroll
      for (int r = 0; r < 8; r++) {
        const int ql = otm * 16 + r + ((lane >> 4) << 3);
        of0[r] *= corr_s[ql];
        of1[r] *= corr_s[ql];
      }
      for (int k2 = 0; k2 < 64; k2 += 32) {
        AFrag af;
        const int am = otm * 16 + (lane & 15);
        const int kof = (lane >> 4) * 16;
        af.q[0] = *(const uint4*)&kT[am][k2 + kof];
        af.q[1] = *(const uint4*)&kT[am][k2 + kof + 8];
        const int bn = lane & 15;
        const int kp0 = (k2 >> 1) + (lane >> 4) * 8;
        BFrag bf0, bf1;
#pragma unroll
        for (int j = 0; j < 8; j++) {
          bf0.u[j] = vsP[kp0 + j][otn0 * 16 + bn];
          bf1.u[j] = vsP[kp0 + j][(otn0 + 1) * 16 + bn];
        }
        of0 = WMMA16(af.v, bf0.v, of0);
        of1 = WMMA16(af.v, bf1.v, of1);
      }
    }
    __syncthreads();
  }
  // write o (divide by softmax denominator)
#pragma unroll
  for (int st = 0; st < 2; st++) {
#pragma unroll
    for (int r = 0; r < 8; r++) {
      const int ql = otm * 16 + r + ((lane >> 4) << 3);
      const int dl = (otn0 + st) * 16 + (lane & 15);
      const float v = (st ? of1[r] : of0[r]) / l_s[ql];
      o16[((size_t)b * 64 + ql) * 768 + d0 + dl] = (_Float16)v;
    }
  }
}

// ---------------------------------------------------------------------------
// Fused assign-attention: scores = aqk @ xn^T + const_g; per-token argmax over
// the 64 groups; scatter-sum raw xn rows into S[b,g,:] and counts into cnt.
// xn^T tile is staged K-paired directly from uint4 global reads.
// ---------------------------------------------------------------------------
__global__ __launch_bounds__(256) void k_assign(const _Float16* __restrict__ aqk,
                                                const _Float16* __restrict__ xn16,
                                                const float* __restrict__ constg,
                                                float* __restrict__ S,
                                                float* __restrict__ cnt) {
  __shared__ __align__(16) _Float16 As[64][40];
  __shared__ unsigned int BsP[16][132];
  __shared__ float sc[64][129];
  __shared__ int gidx[128];
  __shared__ float cloc[64];
  const int b = blockIdx.y;
  const int n0 = blockIdx.x * 128;
  const int tid = threadIdx.x, lane = tid & 31, wave = tid >> 5;
  if (tid < 64) cloc[tid] = 0.f;
  v8f acc[4] = {};
  for (int kk = 0; kk < 768; kk += 32) {
    {
      const int r = tid >> 2, c0 = (tid & 3) * 8;
      *(uint4*)&As[r][c0] = *(const uint4*)(aqk + ((size_t)b * 64 + r) * 768 + kk + c0);
    }
    {
      const int n = tid >> 1, k0h = (tid & 1) * 16;  // halves
      const _Float16* src = xn16 + ((size_t)b * 4096 + n0 + n) * 768 + kk + k0h;
      const uint4 a = *(const uint4*)src;
      const uint4 bq = *(const uint4*)(src + 8);
      const int kps = k0h >> 1;
      BsP[kps + 0][n] = a.x;
      BsP[kps + 1][n] = a.y;
      BsP[kps + 2][n] = a.z;
      BsP[kps + 3][n] = a.w;
      BsP[kps + 4][n] = bq.x;
      BsP[kps + 5][n] = bq.y;
      BsP[kps + 6][n] = bq.z;
      BsP[kps + 7][n] = bq.w;
    }
    __syncthreads();
    const int bn = lane & 15;
    const int kp0 = (lane >> 4) * 8;
    BFrag bf;
#pragma unroll
    for (int j = 0; j < 8; j++) bf.u[j] = BsP[kp0 + j][wave * 16 + bn];
#pragma unroll
    for (int t = 0; t < 4; t++) {
      AFrag af;
      const int am = t * 16 + (lane & 15);
      const int kof = (lane >> 4) * 16;
      af.q[0] = *(const uint4*)&As[am][kof];
      af.q[1] = *(const uint4*)&As[am][kof + 8];
      acc[t] = WMMA16(af.v, bf.v, acc[t]);
    }
    __syncthreads();
  }
#pragma unroll
  for (int t = 0; t < 4; t++)
#pragma unroll
    for (int r = 0; r < 8; r++) {
      const int gl = t * 16 + r + ((lane >> 4) << 3);
      const int nl = wave * 16 + (lane & 15);
      sc[gl][nl] = acc[t][r] + constg[b * 64 + gl];
    }
  __syncthreads();
  if (tid < 128) {
    float best = -1e30f;
    int bg = 0;
    for (int g = 0; g < 64; g++) {
      float v = sc[g][tid];
      if (v > best) { best = v; bg = g; }
    }
    gidx[tid] = bg;
    atomicAdd(&cloc[bg], 1.f);
  }
  __syncthreads();
  for (int nn = 0; nn < 128; nn++) {
    const int g = gidx[nn];
    const _Float16* xr = xn16 + ((size_t)b * 4096 + n0 + nn) * 768;
    float* Sr = S + ((size_t)b * 64 + g) * 768;
    for (int c = tid; c < 768; c += 256) atomicAdd(&Sr[c], (float)xr[c]);
  }
  if (tid < 64) atomicAdd(&cnt[b * 64 + tid], cloc[tid]);
}

// ---------------------------------------------------------------------------
// Host-side pipeline
// ---------------------------------------------------------------------------
extern "C" void kernel_launch(void* const* d_in, const int* in_sizes, int n_in,
                              void* d_out, int out_size, void* d_ws, size_t ws_size,
                              hipStream_t stream) {
  (void)in_sizes; (void)n_in; (void)out_size; (void)ws_size;
  constexpr int B_ = 32, N_ = 4096, C_ = 768, H_ = 12, GIN = 128, GOUT = 64;
  constexpr int THID = 384, MHID = 3072, CHID = 3072;

  const float* x          = (const float*)d_in[0];
  const float* gtoks      = (const float*)d_in[1];
  const float* ln_tok_g   = (const float*)d_in[2];
  const float* ln_tok_b   = (const float*)d_in[3];
  const float* ln_x_g     = (const float*)d_in[4];
  const float* ln_x_b     = (const float*)d_in[5];
  const float* inter_w1   = (const float*)d_in[6];
  const float* inter_b1   = (const float*)d_in[7];
  const float* inter_w2   = (const float*)d_in[8];
  const float* inter_b2   = (const float*)d_in[9];
  const float* ln_pt_g    = (const float*)d_in[10];
  const float* ln_pt_b    = (const float*)d_in[11];
  const float* ca_qw      = (const float*)d_in[12];
  const float* ca_qb      = (const float*)d_in[13];
  const float* ca_kw      = (const float*)d_in[14];
  const float* ca_kb      = (const float*)d_in[15];
  const float* ca_vw      = (const float*)d_in[16];
  const float* ca_vb      = (const float*)d_in[17];
  const float* ca_pw      = (const float*)d_in[18];
  const float* ca_pb      = (const float*)d_in[19];
  const float* ca_ln2_g   = (const float*)d_in[20];
  const float* ca_ln2_b   = (const float*)d_in[21];
  const float* ca_m1w     = (const float*)d_in[22];
  const float* ca_m1b     = (const float*)d_in[23];
  const float* ca_m2w     = (const float*)d_in[24];
  const float* ca_m2b     = (const float*)d_in[25];
  const float* ca_lnp_g   = (const float*)d_in[26];
  const float* ca_lnp_b   = (const float*)d_in[27];
  const float* as_qw      = (const float*)d_in[28];
  const float* as_qb      = (const float*)d_in[29];
  const float* as_kw      = (const float*)d_in[30];
  const float* as_kb      = (const float*)d_in[31];
  const float* as_vw      = (const float*)d_in[32];
  const float* as_vb      = (const float*)d_in[33];
  const float* as_pw      = (const float*)d_in[34];
  const float* as_pb      = (const float*)d_in[35];
  const float* ln_nx_g    = (const float*)d_in[36];
  const float* ln_nx_b    = (const float*)d_in[37];
  const float* mc_w1      = (const float*)d_in[38];
  const float* mc_b1      = (const float*)d_in[39];
  const float* mc_w2      = (const float*)d_in[40];
  const float* mc_b2      = (const float*)d_in[41];

  char* wsp = (char*)d_ws;
  size_t off = 0;
  auto alloc = [&](size_t bytes) -> void* {
    void* p = wsp + off;
    off = (off + bytes + 255) & ~(size_t)255;
    return p;
  };
  auto A16 = [&](size_t n) { return (_Float16*)alloc(n * sizeof(_Float16)); };
  auto A32 = [&](size_t n) { return (float*)alloc(n * sizeof(float)); };

  _Float16* xn16    = A16((size_t)B_ * N_ * C_);
  _Float16* gtT16   = A16((size_t)B_ * C_ * GIN);
  _Float16* w_iw1   = A16((size_t)GIN * THID);
  _Float16* w_iw2   = A16((size_t)THID * GOUT);
  _Float16* w_qw    = A16((size_t)C_ * C_);
  _Float16* w_kw    = A16((size_t)C_ * C_);
  _Float16* w_vw    = A16((size_t)C_ * C_);
  _Float16* w_pw    = A16((size_t)C_ * C_);
  _Float16* w_m1    = A16((size_t)C_ * MHID);
  _Float16* w_m2    = A16((size_t)MHID * C_);
  _Float16* w_asq   = A16((size_t)C_ * C_);
  _Float16* w_askT  = A16((size_t)C_ * C_);
  _Float16* w_asv   = A16((size_t)C_ * C_);
  _Float16* w_asp   = A16((size_t)C_ * C_);
  _Float16* w_mc1   = A16((size_t)C_ * CHID);
  _Float16* w_mc2   = A16((size_t)CHID * C_);
  _Float16* t1_16   = A16((size_t)B_ * C_ * THID);
  float*    t2_32   = A32((size_t)B_ * C_ * GOUT);
  _Float16* pgt16   = A16((size_t)B_ * GOUT * C_);
  float*    pgt32   = A32((size_t)B_ * GOUT * C_);
  _Float16* q16     = A16((size_t)B_ * GOUT * C_);
  _Float16* o16     = A16((size_t)B_ * GOUT * C_);
  float*    y32     = A32((size_t)B_ * GOUT * C_);
  _Float16* yln16   = A16((size_t)B_ * GOUT * C_);
  _Float16* h1_16   = A16((size_t)B_ * GOUT * MHID);
  float*    y2_32   = A32((size_t)B_ * GOUT * C_);
  _Float16* pgt2_16 = A16((size_t)B_ * GOUT * C_);
  float*    pgt2_32 = A32((size_t)B_ * GOUT * C_);
  _Float16* aq16    = A16((size_t)B_ * GOUT * C_);
  _Float16* aqk16   = A16((size_t)B_ * GOUT * C_);
  float*    constg  = A32((size_t)B_ * GOUT);
  float*    S32     = A32((size_t)B_ * GOUT * C_);
  float*    cnt     = A32((size_t)B_ * GOUT);
  _Float16* S16     = A16((size_t)B_ * GOUT * C_);
  _Float16* T1_16   = A16((size_t)B_ * GOUT * C_);
  float*    nx32    = A32((size_t)B_ * GOUT * C_);
  _Float16* nxln16  = A16((size_t)B_ * GOUT * C_);
  _Float16* h2_16   = A16((size_t)B_ * GOUT * CHID);

  auto cvt = [&](const float* in, _Float16* out, long long n) {
    int blocks = (int)((n + 255) / 256);
    if (blocks > 16384) blocks = 16384;
    k_cvt16<<<blocks, 256, 0, stream>>>(in, out, n);
  };

  // ---- weight conversions to f16 ----
  cvt(inter_w1, w_iw1, (long long)GIN * THID);
  cvt(inter_w2, w_iw2, (long long)THID * GOUT);
  cvt(ca_qw, w_qw, (long long)C_ * C_);
  cvt(ca_kw, w_kw, (long long)C_ * C_);
  cvt(ca_vw, w_vw, (long long)C_ * C_);
  cvt(ca_pw, w_pw, (long long)C_ * C_);
  cvt(ca_m1w, w_m1, (long long)C_ * MHID);
  cvt(ca_m2w, w_m2, (long long)MHID * C_);
  cvt(as_qw, w_asq, (long long)C_ * C_);
  k_transpose16<<<4608, 256, 0, stream>>>(as_kw, w_askT, C_, C_);
  cvt(as_vw, w_asv, (long long)C_ * C_);
  cvt(as_pw, w_asp, (long long)C_ * C_);
  cvt(mc_w1, w_mc1, (long long)C_ * CHID);
  cvt(mc_w2, w_mc2, (long long)CHID * C_);

  // ---- norms of the inputs ----
  k_ln768<<<B_ * N_, 256, 0, stream>>>(x, ln_x_g, ln_x_b, xn16, nullptr);
  k_ln_gt_T<<<dim3(GIN, B_), 256, 0, stream>>>(gtoks, ln_tok_g, ln_tok_b, gtT16);

  // ---- project_group_token (Mlp over token dim) ----
  k_gemm<<<dim3(THID / 64, C_ / 64, B_), 256, 0, stream>>>(
      gtT16, w_iw1, inter_b1, nullptr, nullptr, t1_16, nullptr, C_, THID, GIN, 1.f, 1);
  k_gemm<<<dim3(GOUT / 64, C_ / 64, B_), 256, 0, stream>>>(
      t1_16, w_iw2, inter_b2, nullptr, t2_32, nullptr, nullptr, C_, GOUT, THID, 1.f, 0);
  k_ln_t2T<<<dim3(GOUT, B_), 256, 0, stream>>>(t2_32, ln_pt_g, ln_pt_b, pgt16, pgt32);

  // ---- cross-attention (pre_assign_attn) ----
  k_gemm<<<dim3(C_ / 64, 1, B_), 256, 0, stream>>>(
      pgt16, w_qw, ca_qb, nullptr, nullptr, q16, nullptr, GOUT, C_, C_, 0.125f, 0);
  k_flash<<<dim3(H_, B_), 256, 0, stream>>>(q16, xn16, w_kw, w_vw, ca_kb, ca_vb, o16);
  k_gemm<<<dim3(C_ / 64, 1, B_), 256, 0, stream>>>(
      o16, w_pw, ca_pb, pgt32, y32, nullptr, nullptr, GOUT, C_, C_, 1.f, 0);
  k_ln768<<<B_ * GOUT, 256, 0, stream>>>(y32, ca_ln2_g, ca_ln2_b, yln16, nullptr);
  k_gemm<<<dim3(MHID / 64, 1, B_), 256, 0, stream>>>(
      yln16, w_m1, ca_m1b, nullptr, nullptr, h1_16, nullptr, GOUT, MHID, C_, 1.f, 1);
  k_gemm<<<dim3(C_ / 64, 1, B_), 256, 0, stream>>>(
      h1_16, w_m2, ca_m2b, y32, y2_32, nullptr, nullptr, GOUT, C_, MHID, 1.f, 0);
  k_ln768<<<B_ * GOUT, 256, 0, stream>>>(y2_32, ca_lnp_g, ca_lnp_b, pgt2_16, pgt2_32);

  // ---- assign attention (hard one-hot, algebraically fused) ----
  const float ascale = 0.03608439182435161f;  // 768^-0.5
  k_gemm<<<dim3(C_ / 64, 1, B_), 256, 0, stream>>>(
      pgt2_16, w_asq, as_qb, nullptr, nullptr, aq16, nullptr, GOUT, C_, C_, ascale, 0);
  k_gemm<<<dim3(C_ / 64, 1, B_), 256, 0, stream>>>(
      aq16, w_askT, nullptr, nullptr, nullptr, aqk16, nullptr, GOUT, C_, C_, 1.f, 0);
  k_rowdot768<<<B_ * GOUT, 256, 0, stream>>>(aq16, as_kb, constg);
  k_zero<<<6144, 256, 0, stream>>>(S32, (long long)B_ * GOUT * C_);
  k_zero<<<8, 256, 0, stream>>>(cnt, (long long)B_ * GOUT);
  k_assign<<<dim3(N_ / 128, B_), 256, 0, stream>>>(aqk16, xn16, constg, S32, cnt);
  cvt(S32, S16, (long long)B_ * GOUT * C_);
  // new_x = ((S@Wv + cnt*vb)/(cnt+1)) @ Wp + pb + pgt2
  k_gemm<<<dim3(C_ / 64, 1, B_), 256, 0, stream>>>(
      S16, w_asv, as_vb, nullptr, nullptr, T1_16, cnt, GOUT, C_, C_, 1.f, 0);
  k_gemm<<<dim3(C_ / 64, 1, B_), 256, 0, stream>>>(
      T1_16, w_asp, as_pb, pgt2_32, nx32, nullptr, nullptr, GOUT, C_, C_, 1.f, 0);

  // ---- final channel MLP ----
  k_ln768<<<B_ * GOUT, 256, 0, stream>>>(nx32, ln_nx_g, ln_nx_b, nxln16, nullptr);
  k_gemm<<<dim3(CHID / 64, 1, B_), 256, 0, stream>>>(
      nxln16, w_mc1, mc_b1, nullptr, nullptr, h2_16, nullptr, GOUT, CHID, C_, 1.f, 1);
  k_gemm<<<dim3(C_ / 64, 1, B_), 256, 0, stream>>>(
      h2_16, w_mc2, mc_b2, nx32, (float*)d_out, nullptr, nullptr, GOUT, C_, CHID, 1.f, 0);
}
